// GIN_ogb_10101763080474
// MI455X (gfx1250) — matrix operand order repaired and checked by reading, not compiled, not run on hardware
//
#include <hip/hip_runtime.h>
#include <stdint.h>

#define N_NODES   100000
#define N_EDGES   1600000
#define DIM       64
#define N_GRAPHS  512
#define BN_EPS    1e-5f
#define NN        ((long long)N_NODES * DIM)

typedef __attribute__((ext_vector_type(16))) __bf16 v16bf;
typedef __attribute__((ext_vector_type(8)))  float  v8f;

// round-to-nearest-even f32 -> bf16
__device__ __forceinline__ unsigned short f2bfu(float f) {
  union { float f; unsigned int u; } c; c.f = f;
  unsigned int u = c.u;
  unsigned int r = u + 0x7FFFu + ((u >> 16) & 1u);
  return (unsigned short)(r >> 16);
}
__device__ __forceinline__ __bf16 tobf(float f) {
  return __builtin_bit_cast(__bf16, f2bfu(f));
}

// ---------------------------------------------------------------- zero
__global__ void k_zero(float* __restrict__ p, long long n) {
  long long i = (long long)blockIdx.x * blockDim.x + threadIdx.x;
  if (i < n) p[i] = 0.f;
}

// ------------------------------------------------- weight -> bf16, transposed
// W is [64 (k=in)][64 (n=out)] f32 ; wt is [64 (n)][64 (k)] bf16
__global__ void k_w2bf(const float* __restrict__ W, unsigned short* __restrict__ wt) {
  int i = blockIdx.x * blockDim.x + threadIdx.x;   // 0..4095
  int k = i >> 6, n = i & 63;
  wt[n * DIM + k] = f2bfu(W[i]);
}

// ---------------------------------------------------------------- edge aggregate
// agg[dst] += h[src]   16 threads/edge, float4 each, native f32 atomics (L2)
__global__ void k_aggregate(const float* __restrict__ h, const int* __restrict__ ei,
                            float* __restrict__ agg) {
  long long t = (long long)blockIdx.x * blockDim.x + threadIdx.x;
  int e = (int)(t >> 4);
  if (e >= N_EDGES) return;
  int c = ((int)t & 15) * 4;
  int src = ei[e];
  int dst = ei[N_EDGES + e];
  const float4 v = *(const float4*)(h + (long long)src * DIM + c);
  float* p = agg + (long long)dst * DIM + c;
  unsafeAtomicAdd(p + 0, v.x);
  unsafeAtomicAdd(p + 1, v.y);
  unsafeAtomicAdd(p + 2, v.z);
  unsafeAtomicAdd(p + 3, v.w);
}

// ---------------------------------------------------------------- pooled += src (by graph)
__global__ void k_pool(const float* __restrict__ src, const int* __restrict__ batch,
                       float* __restrict__ pooled) {
  long long t = (long long)blockIdx.x * blockDim.x + threadIdx.x;
  int row = (int)(t >> 4);
  if (row >= N_NODES) return;
  int c = ((int)t & 15) * 4;
  float4 v = *(const float4*)(src + (long long)row * DIM + c);
  float* pp = pooled + (long long)batch[row] * DIM + c;
  unsafeAtomicAdd(pp + 0, v.x);
  unsafeAtomicAdd(pp + 1, v.y);
  unsafeAtomicAdd(pp + 2, v.z);
  unsafeAtomicAdd(pp + 3, v.w);
}

// ---------------------------------------------------------------- WMMA GEMM [nrows,64] x [64,64]
// mode 0: A = in0 + in1            (GIN sum)
// mode 1: A = relu(in0*sc + sh)    (fused BN+ReLU on input)
// epilogue: out = A@W + bias ; per-column sum/sumsq accumulated for BN stats.
// 256 threads = 8 waves, wave w owns output rows [r0+16w, r0+16w+16), all 64 cols.
__global__ void __launch_bounds__(256)
k_gemm64(const float* __restrict__ in0, const float* __restrict__ in1,
         const unsigned short* __restrict__ wt,           // [n][k] bf16
         const float* __restrict__ sc, const float* __restrict__ sh,
         const float* __restrict__ bias,
         float* __restrict__ out, float* __restrict__ gsum, float* __restrict__ gsq,
         int nrows, int mode) {
  __shared__ float ssum[DIM], ssq[DIM];
  const int tid  = threadIdx.x;
  if (tid < DIM) { ssum[tid] = 0.f; ssq[tid] = 0.f; }
  const int wave = tid >> 5;
  const int lane = tid & 31;
  const int half = lane >> 4;          // 0 | 1
  const int l16  = lane & 15;
  const int r0   = blockIdx.x * 128 + wave * 16;

  v8f acc[4];
  const v8f zed = {0.f,0.f,0.f,0.f,0.f,0.f,0.f,0.f};
#pragma unroll
  for (int nt = 0; nt < 4; ++nt) acc[nt] = zed;

  const int  arow   = r0 + l16;
  const bool avalid = (arow < nrows);
  const long long abase = (long long)(avalid ? arow : 0) * DIM;

#pragma unroll
  for (int kt = 0; kt < 2; ++kt) {
    const int kb = kt * 32 + half * 16;      // A K-base (contiguous, matches B)
    v16bf afrag;
#pragma unroll
    for (int g = 0; g < 2; ++g) {            // g=0: K=kb..kb+7 ; g=1: K=kb+8..kb+15
      const int k0 = kb + g * 8;
      float4 p0 = *(const float4*)(in0 + abase + k0);
      float4 p1 = *(const float4*)(in0 + abase + k0 + 4);
      float v[8] = {p0.x,p0.y,p0.z,p0.w,p1.x,p1.y,p1.z,p1.w};
      if (mode == 0) {
        float4 q0 = *(const float4*)(in1 + abase + k0);
        float4 q1 = *(const float4*)(in1 + abase + k0 + 4);
        float a[8] = {q0.x,q0.y,q0.z,q0.w,q1.x,q1.y,q1.z,q1.w};
#pragma unroll
        for (int e = 0; e < 8; ++e) v[e] += a[e];
      } else {
        float4 s0 = *(const float4*)(sc + k0);
        float4 s1 = *(const float4*)(sc + k0 + 4);
        float4 t0 = *(const float4*)(sh + k0);
        float4 t1 = *(const float4*)(sh + k0 + 4);
        float s[8] = {s0.x,s0.y,s0.z,s0.w,s1.x,s1.y,s1.z,s1.w};
        float t[8] = {t0.x,t0.y,t0.z,t0.w,t1.x,t1.y,t1.z,t1.w};
#pragma unroll
        for (int e = 0; e < 8; ++e) v[e] = fmaxf(fmaf(v[e], s[e], t[e]), 0.f);
      }
      if (!avalid) {
#pragma unroll
        for (int e = 0; e < 8; ++e) v[e] = 0.f;
      }
#pragma unroll
      for (int e = 0; e < 8; ++e) afrag[g * 8 + e] = tobf(v[e]);
    }
#pragma unroll
    for (int nt = 0; nt < 4; ++nt) {
      const int n   = nt * 16 + l16;
      const int kb2 = kt * 32 + half * 16;   // B K-base (contiguous, matches A)
      v16bf bfrag = *(const v16bf*)(wt + n * DIM + kb2);
      acc[nt] = __builtin_amdgcn_wmma_f32_16x16x32_bf16(
          false, afrag, false, bfrag, (short)0, acc[nt], false, false);
    }
  }

  __syncthreads();
#pragma unroll
  for (int nt = 0; nt < 4; ++nt) {
    const int col = nt * 16 + l16;
    const float b = bias[col];
    float lsum = 0.f, lsq = 0.f;
#pragma unroll
    for (int j = 0; j < 8; ++j) {
      const int row = r0 + half * 8 + j;     // C/D layout: VGPR j, lane-half selects M/M+8
      float val = acc[nt][j] + b;
      if (row < nrows) {
        out[(long long)row * DIM + col] = val;
        lsum += val;
        lsq  += val * val;
      }
    }
    atomicAdd(&ssum[col], lsum);
    atomicAdd(&ssq[col],  lsq);
  }
  __syncthreads();
  if (tid < DIM) {
    unsafeAtomicAdd(&gsum[tid], ssum[tid]);
    unsafeAtomicAdd(&gsq[tid],  ssq[tid]);
  }
}

// ---------------------------------------------------------------- BN stats -> scale/shift
__global__ void k_bnfinal(const float* __restrict__ gsum, const float* __restrict__ gsq,
                          const float* __restrict__ g, const float* __restrict__ b,
                          float* __restrict__ sc, float* __restrict__ sh, float inv_n) {
  int c = threadIdx.x;
  if (c >= DIM) return;
  float mu  = gsum[c] * inv_n;
  float var = gsq[c] * inv_n - mu * mu;
  float s   = g[c] * rsqrtf(var + BN_EPS);
  sc[c] = s;
  sh[c] = b[c] - mu * s;
}

// ---------------------------------------------------------------- hnext = relu(bn(z)); pooled += hnext
__global__ void k_apply_pool(const float* __restrict__ z, const float* __restrict__ sc,
                             const float* __restrict__ sh, const int* __restrict__ batch,
                             float* __restrict__ hnext, float* __restrict__ pooled) {
  long long t = (long long)blockIdx.x * blockDim.x + threadIdx.x;
  int row = (int)(t >> 4);
  if (row >= N_NODES) return;
  int c = ((int)t & 15) * 4;
  float4 v = *(const float4*)(z  + (long long)row * DIM + c);
  float4 s = *(const float4*)(sc + c);
  float4 b = *(const float4*)(sh + c);
  float4 r;
  r.x = fmaxf(fmaf(v.x, s.x, b.x), 0.f);
  r.y = fmaxf(fmaf(v.y, s.y, b.y), 0.f);
  r.z = fmaxf(fmaf(v.z, s.z, b.z), 0.f);
  r.w = fmaxf(fmaf(v.w, s.w, b.w), 0.f);
  *(float4*)(hnext + (long long)row * DIM + c) = r;
  float* pp = pooled + (long long)batch[row] * DIM + c;
  unsafeAtomicAdd(pp + 0, r.x);
  unsafeAtomicAdd(pp + 1, r.y);
  unsafeAtomicAdd(pp + 2, r.z);
  unsafeAtomicAdd(pp + 3, r.w);
}

// ---------------------------------------------------------------- readout: out = sum_i pooled_i @ Wfc_i + sum_i b_i
__global__ void __launch_bounds__(256)
k_readout(const float* __restrict__ pooled, const unsigned short* __restrict__ wtfc,
          const float* __restrict__ b0, const float* __restrict__ b1,
          const float* __restrict__ b2, const float* __restrict__ b3,
          const float* __restrict__ b4, float* __restrict__ out) {
  const int tid  = threadIdx.x;
  const int wave = tid >> 5;
  const int lane = tid & 31;
  const int half = lane >> 4;
  const int l16  = lane & 15;
  const int r0   = blockIdx.x * 128 + wave * 16;   // 4 blocks x 128 = 512 rows exactly

  v8f acc[4];
  const v8f zed = {0.f,0.f,0.f,0.f,0.f,0.f,0.f,0.f};
#pragma unroll
  for (int nt = 0; nt < 4; ++nt) acc[nt] = zed;

  const long long abase = (long long)(r0 + l16) * DIM;
  for (int i = 0; i < 5; ++i) {
    const float* P = pooled + (long long)i * N_GRAPHS * DIM;
    const unsigned short* Wt = wtfc + i * DIM * DIM;
#pragma unroll
    for (int kt = 0; kt < 2; ++kt) {
      const int kb = kt * 32 + half * 16;
      v16bf afrag;
#pragma unroll
      for (int g = 0; g < 2; ++g) {
        const int k0 = kb + g * 8;
        float4 p0 = *(const float4*)(P + abase + k0);
        float4 p1 = *(const float4*)(P + abase + k0 + 4);
        float v[8] = {p0.x,p0.y,p0.z,p0.w,p1.x,p1.y,p1.z,p1.w};
#pragma unroll
        for (int e = 0; e < 8; ++e) afrag[g * 8 + e] = tobf(v[e]);
      }
#pragma unroll
      for (int nt = 0; nt < 4; ++nt) {
        const int n   = nt * 16 + l16;
        const int kb2 = kt * 32 + half * 16;
        v16bf bfrag = *(const v16bf*)(Wt + n * DIM + kb2);
        acc[nt] = __builtin_amdgcn_wmma_f32_16x16x32_bf16(
            false, afrag, false, bfrag, (short)0, acc[nt], false, false);
      }
    }
  }
#pragma unroll
  for (int nt = 0; nt < 4; ++nt) {
    const int col = nt * 16 + l16;
    const float bb = b0[col] + b1[col] + b2[col] + b3[col] + b4[col];
#pragma unroll
    for (int j = 0; j < 8; ++j) {
      const int row = r0 + half * 8 + j;
      out[(long long)row * DIM + col] = acc[nt][j] + bb;
    }
  }
}

// ================================================================ host
extern "C" void kernel_launch(void* const* d_in, const int* in_sizes, int n_in,
                              void* d_out, int out_size, void* d_ws, size_t ws_size,
                              hipStream_t stream) {
  // inputs in setup_inputs() insertion order:
  // 0:x 1:edge_index 2:batch
  // convs[l]: W1,b1,bn_g,bn_b,W2,b2  -> 3 + 6*l
  // bns[l]:   g,b                    -> 27 + 2*l
  // fcs[i]:   W,b                    -> 35 + 2*i
  const float* x     = (const float*)d_in[0];
  const int*   ei    = (const int*)d_in[1];
  const int*   batch = (const int*)d_in[2];
  const float* cW1[4];  const float* cb1[4]; const float* cbng[4]; const float* cbnb[4];
  const float* cW2[4];  const float* cb2[4];
  for (int l = 0; l < 4; ++l) {
    const int base = 3 + 6 * l;
    cW1[l]  = (const float*)d_in[base + 0];
    cb1[l]  = (const float*)d_in[base + 1];
    cbng[l] = (const float*)d_in[base + 2];
    cbnb[l] = (const float*)d_in[base + 3];
    cW2[l]  = (const float*)d_in[base + 4];
    cb2[l]  = (const float*)d_in[base + 5];
  }
  const float* bng2[4]; const float* bnb2[4];
  for (int l = 0; l < 4; ++l) {
    bng2[l] = (const float*)d_in[27 + 2 * l];
    bnb2[l] = (const float*)d_in[28 + 2 * l];
  }
  const float* fW[5]; const float* fb[5];
  for (int i = 0; i < 5; ++i) {
    fW[i] = (const float*)d_in[35 + 2 * i];
    fb[i] = (const float*)d_in[36 + 2 * i];
  }

  // workspace layout (floats)
  float* agg    = (float*)d_ws;
  float* hbuf   = agg  + NN;
  float* ybuf   = hbuf + NN;
  float* pooled = ybuf + NN;                      // 5 * 512 * 64
  float* gsum   = pooled + 5LL * N_GRAPHS * DIM;  // 64
  float* gsq    = gsum + DIM;                     // 64
  float* sc     = gsq  + DIM;                     // 64
  float* sh     = sc   + DIM;                     // 64
  unsigned short* wt = (unsigned short*)(sh + DIM);  // 13 * 4096 bf16

  // weight prep: slots 0-3 W1, 4-7 W2, 8-12 fc
  for (int l = 0; l < 4; ++l) {
    k_w2bf<<<16, 256, 0, stream>>>(cW1[l], wt + (size_t)l * 4096);
    k_w2bf<<<16, 256, 0, stream>>>(cW2[l], wt + (size_t)(4 + l) * 4096);
  }
  for (int i = 0; i < 5; ++i)
    k_w2bf<<<16, 256, 0, stream>>>(fW[i], wt + (size_t)(8 + i) * 4096);

  const long long nPooled = 5LL * N_GRAPHS * DIM;
  k_zero<<<(unsigned)((nPooled + 255) / 256), 256, 0, stream>>>(pooled, nPooled);
  k_pool<<<(N_NODES * 16 + 255) / 256, 256, 0, stream>>>(x, batch, pooled);

  const float invN = 1.f / (float)N_NODES;
  const int gemmBlocks = (N_NODES + 127) / 128;
  const float* hcur = x;
  for (int l = 0; l < 4; ++l) {
    k_zero<<<(unsigned)((NN + 255) / 256), 256, 0, stream>>>(agg, NN);
    k_aggregate<<<(N_EDGES * 16) / 256, 256, 0, stream>>>(hcur, ei, agg);

    k_zero<<<1, 128, 0, stream>>>(gsum, 128);     // gsum+gsq contiguous
    k_gemm64<<<gemmBlocks, 256, 0, stream>>>(hcur, agg, wt + (size_t)l * 4096,
                                             nullptr, nullptr, cb1[l],
                                             ybuf, gsum, gsq, N_NODES, 0);
    k_bnfinal<<<1, 64, 0, stream>>>(gsum, gsq, cbng[l], cbnb[l], sc, sh, invN);

    k_zero<<<1, 128, 0, stream>>>(gsum, 128);
    // z reuses agg (dead after GEMM1)
    k_gemm64<<<gemmBlocks, 256, 0, stream>>>(ybuf, nullptr, wt + (size_t)(4 + l) * 4096,
                                             sc, sh, cb2[l],
                                             agg, gsum, gsq, N_NODES, 1);
    k_bnfinal<<<1, 64, 0, stream>>>(gsum, gsq, bng2[l], bnb2[l], sc, sh, invN);

    k_apply_pool<<<(N_NODES * 16 + 255) / 256, 256, 0, stream>>>(
        agg, sc, sh, batch, hbuf, pooled + (long long)(1 + l) * N_GRAPHS * DIM);
    hcur = hbuf;
  }

  k_readout<<<4, 256, 0, stream>>>(pooled, wt + (size_t)8 * 4096,
                                   fb[0], fb[1], fb[2], fb[3], fb[4],
                                   (float*)d_out);
}
